// bioxcpath_pooling_67138928771659
// MI455X (gfx1250) — matrix operand order repaired and checked by reading, not compiled, not run on hardware
//
#include <hip/hip_runtime.h>
#include <hip/hip_bf16.h>
#include <math.h>

// ---------------------------------------------------------------------------
// BioXCPath GATv2+SAGPool pipeline for MI455X (gfx1250, wave32, WMMA).
// GEMMs: bf16 V_WMMA_F32_16X16X32_BF16, pre-packed fragment layouts,
// 8 N-tiles per wave register blocking, compile-time-KT fully unrolled
// K-loop + relaxed occupancy so the scheduler pipelines loads across WMMAs.
// ---------------------------------------------------------------------------

#define N_NODES   30000
#define N_EDGES   240000
#define F_IN      1024
#define HID       256
#define HD        512      // HEADS * HID
#define WALK      16
#define S_STAINS  5
#define NLAYERS   4

#define CDIV(a,b) (((a)+(b)-1)/(b))

typedef __attribute__((ext_vector_type(16))) __bf16 v16bf;
typedef __attribute__((ext_vector_type(8)))  float  v8f;

// ---- helpers ---------------------------------------------------------------

__device__ __forceinline__ __bf16 f2bf(float f) {
  unsigned u = __builtin_bit_cast(unsigned, f);
  u += 0x7fffu + ((u >> 16) & 1u);                 // round-to-nearest-even
  unsigned short s = (unsigned short)(u >> 16);
  return __builtin_bit_cast(__bf16, s);
}

__device__ __forceinline__ void atomicMaxF(float* addr, float v) {
  int* ai = (int*)addr;
  int old = __float_as_int(*addr);
  while (__int_as_float(old) < v) {
    int assumed = old;
    old = atomicCAS(ai, assumed, __float_as_int(v));
    if (old == assumed) break;
  }
}

// ---- fills -----------------------------------------------------------------

__global__ void k_fill_f(float* p, float v, int n) {
  int t = blockIdx.x * blockDim.x + threadIdx.x;
  if (t < n) p[t] = v;
}
__global__ void k_fill_i(int* p, int v, int n) {
  int t = blockIdx.x * blockDim.x + threadIdx.x;
  if (t < n) p[t] = v;
}
__global__ void k_iota(int* p, int n) {
  int t = blockIdx.x * blockDim.x + threadIdx.x;
  if (t < n) p[t] = t;
}

// ---- WMMA GEMM: operand packing --------------------------------------------
// A (16x32 bf16, wave32): lanes 0-15 row M=lane, halves 0..7 -> K 0..7,
//   halves 8..15 -> K 16..23; lanes 16-31 same rows, K+8 / K+24.
// Tile order: (mt*KT + kt)*32 + lane, 16 halves per lane (32B contiguous).
__global__ void k_packA(const float* __restrict__ x, const float* __restrict__ rw,
                        v16bf* __restrict__ Ap, int n, int xdim, int indim,
                        int MT, int KT) {
  int gid = blockIdx.x * blockDim.x + threadIdx.x;
  int total = MT * KT * 32;
  if (gid >= total) return;
  int lane = gid & 31;
  int tile = gid >> 5;
  int kt = tile % KT, mt = tile / KT;
  int m = mt * 16 + (lane & 15);
  int kofs = kt * 32 + ((lane & 16) ? 8 : 0);
  v16bf v;
#pragma unroll
  for (int h = 0; h < 16; ++h) {
    int K = kofs + (h < 8 ? h : h + 8);
    float f = 0.f;
    if (m < n && K < indim)
      f = (K < xdim) ? x[(size_t)m * xdim + K] : rw[(size_t)m * WALK + (K - xdim)];
    v[h] = f2bf(f);
  }
  Ap[gid] = v;
}

// B (32x16 bf16): lanes 0-15 col N=lane, halves 0..7 -> K 0..7, 8..15 -> K 16..23;
// lanes 16-31 col N=lane-16, K+8 / K+24. Tile order: (kt*NT + nt)*32 + lane.  NT = 512/16 = 32.
__global__ void k_packB(const float* __restrict__ W, v16bf* __restrict__ Bp,
                        int indim, int KT) {
  int gid = blockIdx.x * blockDim.x + threadIdx.x;
  int total = KT * 32 * 32;
  if (gid >= total) return;
  int lane = gid & 31;
  int tile = gid >> 5;
  int nt = tile & 31, kt = tile >> 5;
  int nn = nt * 16 + (lane & 15);
  int kbase = kt * 32 + ((lane & 16) ? 8 : 0);
  v16bf v;
#pragma unroll
  for (int h = 0; h < 16; ++h) {
    int K = kbase + (h < 8 ? h : h + 8);
    v[h] = f2bf(K < indim ? W[(size_t)K * HD + nn] : 0.f);
  }
  Bp[gid] = v;
}

// C row-major fp32 [Mrows x 512], bias added.  One block (4 waves) covers a
// full 16x512 C strip: wave w owns N-tiles w*8..w*8+7 with 8 accumulators.
// KT is a compile-time constant: the K-loop fully unrolls and, with the
// relaxed register budget (min 1 block/CU), the scheduler can hoist b128
// fragment loads far ahead of their consuming WMMAs (partial loadcnt waits).
template <int KT>
__global__ __launch_bounds__(128, 1)
void k_gemm(const v16bf* __restrict__ Ap, const v16bf* __restrict__ Bp,
            const float* __restrict__ bias, float* __restrict__ C,
            int Mrows) {
  int lane = threadIdx.x & 31;
  int w    = threadIdx.x >> 5;
  int mt   = blockIdx.x;
  v8f acc[8] = {};
  const v16bf* a = Ap + (size_t)mt * KT * 32 + lane;
  const v16bf* b = Bp + (size_t)(w * 8) * 32 + lane;
#pragma unroll
  for (int kt = 0; kt < KT; ++kt) {
    v16bf av = a[(size_t)kt * 32];
    const v16bf* bk = b + (size_t)kt * 1024;          // kt*NT*32, NT=32
    v16bf bv[8];
#pragma unroll
    for (int j = 0; j < 8; ++j) bv[j] = bk[(size_t)j * 32];
#pragma unroll
    for (int j = 0; j < 8; ++j)
      acc[j] = __builtin_amdgcn_wmma_f32_16x16x32_bf16(
          false, av, false, bv[j], (short)0, acc[j], false, false);
  }
  int m0 = mt * 16 + ((lane & 16) ? 8 : 0);
#pragma unroll
  for (int j = 0; j < 8; ++j) {
    int col = (w * 8 + j) * 16 + (lane & 15);
    float bs = bias[col];
#pragma unroll
    for (int r = 0; r < 8; ++r) {
      int row = m0 + r;
      if (row < Mrows) C[(size_t)row * HD + col] = acc[j][r] + bs;
    }
  }
}

// ---- GATv2 attention -------------------------------------------------------
// Wave per edge (incl. implicit self-loops e in [E, E+n)): both heads' logits
// via 32-lane dot reduce; fused atomic segment-max on destination.
__global__ void k_logits(const float* __restrict__ xl, const float* __restrict__ xr,
                         const float* __restrict__ att,
                         const int* __restrict__ src, const int* __restrict__ dst,
                         const int* __restrict__ mask,
                         float* __restrict__ logit, float* __restrict__ mmax,
                         int E, int n) {
  int gw = (blockIdx.x * blockDim.x + threadIdx.x) >> 5;
  int lane = threadIdx.x & 31;
  int E2 = E + n;
  if (gw >= E2) return;
  int s, d, mk;
  if (gw < E) { s = src[gw]; d = dst[gw]; mk = mask[gw]; }
  else        { s = d = gw - E; mk = 1; }
#pragma unroll
  for (int hh = 0; hh < 2; ++hh) {
    const float* pl = xl + (size_t)s * HD + hh * HID;
    const float* pr = xr + (size_t)d * HD + hh * HID;
    const float* pa = att + hh * HID;
    float part = 0.f;
#pragma unroll
    for (int j = 0; j < 8; ++j) {
      int dim = lane + 32 * j;
      float e = pl[dim] + pr[dim];
      e = e > 0.f ? e : 0.2f * e;                 // leaky_relu
      part += pa[dim] * e;
    }
    for (int off = 16; off; off >>= 1) part += __shfl_xor(part, off, 32);
    if (lane == 0) {
      float lg = mk ? part : -1e9f;
      logit[(size_t)gw * 2 + hh] = lg;
      atomicMaxF(&mmax[d * 2 + hh], lg);
    }
  }
}

// exp(logit - mmax[dst]) in place; atomic segment-sum of denominators.
__global__ void k_exps(float* __restrict__ logit, const int* __restrict__ dst,
                       const int* __restrict__ mask, const float* __restrict__ mmax,
                       float* __restrict__ denom, int E, int n) {
  int t = blockIdx.x * blockDim.x + threadIdx.x;
  int E2 = E + n;
  if (t >= E2 * 2) return;
  int e = t >> 1, hh = t & 1;
  int d, mk;
  if (e < E) { d = dst[e]; mk = mask[e]; } else { d = e - E; mk = 1; }
  float ex = mk ? __expf(logit[t] - mmax[d * 2 + hh]) : 0.f;
  logit[t] = ex;
  if (ex != 0.f) atomicAdd(&denom[d * 2 + hh], ex);
}

// Wave per edge: agg[d] += alpha * xl[s] (both heads, 512 dims).
__global__ void k_agg(const float* __restrict__ xl, const float* __restrict__ ex,
                      const float* __restrict__ denom,
                      const int* __restrict__ src, const int* __restrict__ dst,
                      float* __restrict__ agg, int E, int n) {
  int gw = (blockIdx.x * blockDim.x + threadIdx.x) >> 5;
  int lane = threadIdx.x & 31;
  int E2 = E + n;
  if (gw >= E2) return;
  int s, d;
  if (gw < E) { s = src[gw]; d = dst[gw]; } else { s = d = gw - E; }
  float a0 = ex[(size_t)gw * 2 + 0];
  float a1 = ex[(size_t)gw * 2 + 1];
  if (a0 == 0.f && a1 == 0.f) return;             // masked edge: contributes 0
  a0 /= denom[d * 2 + 0];
  a1 /= denom[d * 2 + 1];
#pragma unroll
  for (int j = 0; j < 16; ++j) {
    int dim = lane + 32 * j;
    float al = (dim < HID) ? a0 : a1;
    float v = al * xl[(size_t)s * HD + dim];
    if (v != 0.f) atomicAdd(&agg[(size_t)d * HD + dim], v);
  }
}

// x = relu(mean over heads + bias)
__global__ void k_relumean(const float* __restrict__ agg, const float* __restrict__ bias,
                           float* __restrict__ x, int n) {
  int t = blockIdx.x * blockDim.x + threadIdx.x;
  if (t >= n * HID) return;
  int node = t >> 8, dim = t & 255;
  float v = 0.5f * (agg[(size_t)node * HD + dim] + agg[(size_t)node * HD + HID + dim])
          + bias[dim];
  x[t] = v > 0.f ? v : 0.f;
}

// ---- SAGPool score (GraphConv + tanh) --------------------------------------

__global__ void k_msg(const float* __restrict__ x, const int* __restrict__ src,
                      const int* __restrict__ dst, const int* __restrict__ mask,
                      float* __restrict__ msg, int E) {
  int gw = (blockIdx.x * blockDim.x + threadIdx.x) >> 5;
  int lane = threadIdx.x & 31;
  if (gw >= E) return;
  if (!mask[gw]) return;
  int s = src[gw], d = dst[gw];
#pragma unroll
  for (int j = 0; j < 8; ++j) {
    int dim = lane + 32 * j;
    atomicAdd(&msg[(size_t)d * HID + dim], x[(size_t)s * HID + dim]);
  }
}

__global__ void k_score(const float* __restrict__ x, const float* __restrict__ msg,
                        const float* __restrict__ Wrel, const float* __restrict__ brel,
                        const float* __restrict__ Wroot,
                        float* __restrict__ score, unsigned* __restrict__ keys, int n) {
  int gw = (blockIdx.x * blockDim.x + threadIdx.x) >> 5;
  int lane = threadIdx.x & 31;
  if (gw >= n) return;
  float part = 0.f;
#pragma unroll
  for (int j = 0; j < 8; ++j) {
    int dim = lane + 32 * j;
    part += msg[(size_t)gw * HID + dim] * Wrel[dim]
          + x[(size_t)gw * HID + dim] * Wroot[dim];
  }
  for (int off = 16; off; off >>= 1) part += __shfl_xor(part, off, 32);
  if (lane == 0) {
    float sc = tanhf(part + brel[0]);
    score[gw] = sc;
    unsigned u = __builtin_bit_cast(unsigned, sc);
    keys[gw] = (u & 0x80000000u) ? ~u : (u | 0x80000000u);   // order-preserving
  }
}

// ---- device-resident radix top-k select ------------------------------------
// st[0]=prefix  st[1]=kRemaining  st[2]=cntGt  st[3]=cntEq

__global__ void k_init_state(unsigned* st, int k) {
  st[0] = 0u; st[1] = (unsigned)k; st[2] = 0u; st[3] = 0u;
}
__global__ void k_hist(const unsigned* __restrict__ keys, const unsigned* __restrict__ st,
                       int* __restrict__ hist, int n, int shift) {
  int t = blockIdx.x * blockDim.x + threadIdx.x;
  if (t >= n) return;
  unsigned key = keys[t];
  unsigned m = (shift == 24) ? 0u : (0xFFFFFFFFu << (shift + 8));
  if ((key & m) == st[0]) atomicAdd(&hist[(key >> shift) & 255], 1);
}
__global__ void k_scanpick(unsigned* st, const int* __restrict__ hist, int shift) {
  unsigned kRem = st[1], prefix = st[0];
  for (int b = 255; b >= 0; --b) {
    unsigned c = (unsigned)hist[b];
    if (c >= kRem) { st[0] = prefix | ((unsigned)b << shift); break; }
    kRem -= c;
  }
  st[1] = kRem;
}
__global__ void k_compact_gt(const unsigned* __restrict__ keys, const float* __restrict__ score,
                             unsigned* st, int* __restrict__ perm, float* __restrict__ vals,
                             int n) {
  int t = blockIdx.x * blockDim.x + threadIdx.x;
  if (t >= n) return;
  if (keys[t] > st[0]) {
    int p = atomicAdd((int*)&st[2], 1);
    perm[p] = t; vals[p] = score[t];
  }
}
__global__ void k_compact_eq(const unsigned* __restrict__ keys, const float* __restrict__ score,
                             unsigned* st, int* __restrict__ perm, float* __restrict__ vals,
                             int n, int k) {
  int t = blockIdx.x * blockDim.x + threadIdx.x;
  if (t >= n) return;
  if (keys[t] == st[0]) {
    int p = (int)st[2] + atomicAdd((int*)&st[3], 1);
    if (p < k) { perm[p] = t; vals[p] = score[t]; }
  }
}

// ---- gather / remap --------------------------------------------------------

__global__ void k_gather_x(const float* __restrict__ xo, const int* __restrict__ perm,
                           const float* __restrict__ vals, float* __restrict__ xn, int k) {
  int t = blockIdx.x * blockDim.x + threadIdx.x;
  if (t >= k * HID) return;
  int j = t >> 8, dim = t & 255;
  xn[t] = xo[(size_t)perm[j] * HID + dim] * vals[j];
}
__global__ void k_gather_misc(const int* __restrict__ perm, const float* __restrict__ rwo,
                              const int* __restrict__ origo, const int* __restrict__ nattr,
                              float* __restrict__ rwn, int* __restrict__ orign,
                              int* __restrict__ stain, int k) {
  int t = blockIdx.x * blockDim.x + threadIdx.x;
  if (t >= k) return;
  int p = perm[t];
  int o = origo[p];
  orign[t] = o;
  stain[t] = nattr[o];
#pragma unroll
  for (int j = 0; j < WALK; ++j) rwn[(size_t)t * WALK + j] = rwo[(size_t)p * WALK + j];
}
__global__ void k_scatter_inv(const int* __restrict__ perm, int* __restrict__ inv, int k) {
  int t = blockIdx.x * blockDim.x + threadIdx.x;
  if (t < k) inv[perm[t]] = t;
}
__global__ void k_remap(int* __restrict__ src, int* __restrict__ dst, int* __restrict__ mask,
                        const int* __restrict__ inv, int E) {
  int t = blockIdx.x * blockDim.x + threadIdx.x;
  if (t >= E) return;
  int mk = mask[t];
  int ns = inv[src[t]], nd = inv[dst[t]];
  mk = (mk && ns >= 0 && nd >= 0) ? 1 : 0;
  mask[t] = mk;
  src[t] = mk ? ns : 0;
  dst[t] = mk ? nd : 0;
}

// ---- stain-grouped pooling -------------------------------------------------

__global__ void k_pool_scalar(const float* __restrict__ vals, const int* __restrict__ stain,
                              float* __restrict__ wsum, float* __restrict__ cnt,
                              float* __restrict__ sumv, int k) {
  int t = blockIdx.x * blockDim.x + threadIdx.x;
  if (t >= k) return;
  float v = vals[t]; int s = stain[t];
  atomicAdd(&wsum[s], v);
  atomicAdd(&cnt[s], 1.f);
  atomicAdd(sumv, v);
}
__global__ void k_pool_vec(const float* __restrict__ xn, const int* __restrict__ stain,
                           float* __restrict__ avg, float* __restrict__ mx, int k) {
  int t = blockIdx.x * blockDim.x + threadIdx.x;
  if (t >= k * HID) return;
  int j = t >> 8, dim = t & 255;
  int s = stain[j];
  float v = xn[t];
  atomicAdd(&avg[s * HID + dim], v);
  atomicMaxF(&mx[s * HID + dim], v);
}
__global__ void k_finalize(const float* __restrict__ wsum, const float* __restrict__ cnt,
                           const float* __restrict__ avg, const float* __restrict__ mx,
                           const float* __restrict__ sumv, float* __restrict__ out) {
  int dim = threadIdx.x;
  if (dim >= HID) return;
  float sv = *sumv;
  float a = 0.f, m = 0.f;
#pragma unroll
  for (int s = 0; s < S_STAINS; ++s) {
    float w = wsum[s] / sv;
    float c = cnt[s];
    a += w * (avg[s * HID + dim] / fmaxf(c, 1.f));
    m += w * (c > 0.f ? mx[s * HID + dim] : 0.f);
  }
  out[dim] = a;
  out[HID + dim] = m;
}

// ---------------------------------------------------------------------------

extern "C" void kernel_launch(void* const* d_in, const int* in_sizes, int n_in,
                              void* d_out, int out_size, void* d_ws, size_t ws_size,
                              hipStream_t stream) {
  if (n_in < 40) return;
  const float* x_in  = (const float*)d_in[0];
  const float* rw_in = (const float*)d_in[1];
  const int*   eidx  = (const int*)d_in[38];
  const int*   nattr = (const int*)d_in[39];

  // Param leaf ordering: dict-insertion (Wl,bl,Wr,br,att,bias,Wrel,brel,Wroot)
  // vs pytree-sorted (Wl,Wr,Wrel,Wroot,att,bias,bl,br,brel). Disambiguate by
  // in_sizes[3]: 512 => insertion (bl), else => sorted (Wr).
  int iWl, iBl, iWr, iBr, iAtt, iBias, iWrel, iBrel, iWroot;
  if (in_sizes[3] == 512) { iWl=0; iBl=1; iWr=2; iBr=3; iAtt=4; iBias=5; iWrel=6; iBrel=7; iWroot=8; }
  else                    { iWl=0; iWr=1; iWrel=2; iWroot=3; iAtt=4; iBias=5; iBl=6; iBr=7; iBrel=8; }

  // --- workspace carve (max sizes are layer-0 sizes) ---
  char* WS = (char*)d_ws;
  size_t off = 0;
  auto ALLOC = [&](size_t bytes) -> void* {
    void* p = WS + off; off = (off + bytes + 255) & ~(size_t)255; return p;
  };
  float* xMid  = (float*)ALLOC((size_t)N_NODES * HID * 4);     // post-GAT features
  float* xSel  = (float*)ALLOC((size_t)N_NODES * HID * 4);     // pooled features
  float* rwA   = (float*)ALLOC((size_t)N_NODES * WALK * 4);
  float* rwB   = (float*)ALLOC((size_t)N_NODES * WALK * 4);
  int*   origA = (int*)ALLOC((size_t)N_NODES * 4);
  int*   origB = (int*)ALLOC((size_t)N_NODES * 4);
  int*   srcB  = (int*)ALLOC((size_t)N_EDGES * 4);
  int*   dstB  = (int*)ALLOC((size_t)N_EDGES * 4);
  int*   maskB = (int*)ALLOC((size_t)N_EDGES * 4);
  v16bf* Apack = (v16bf*)ALLOC((size_t)1875 * 33 * 1024);      // MT*KT tiles * 1KB
  v16bf* Bpack = (v16bf*)ALLOC((size_t)33 * 32 * 1024);
  float* xl    = (float*)ALLOC((size_t)N_NODES * HD * 4);
  float* xr    = (float*)ALLOC((size_t)N_NODES * HD * 4);
  float* agg   = (float*)ALLOC((size_t)N_NODES * HD * 4);
  float* logit = (float*)ALLOC((size_t)(N_EDGES + N_NODES) * 2 * 4);
  float* mmax  = (float*)ALLOC((size_t)N_NODES * 2 * 4);
  float* denom = (float*)ALLOC((size_t)N_NODES * 2 * 4);
  // dead-buffer aliases (phases are disjoint in stream order):
  float*    msg   = agg;                                       // agg dead after relumean
  float*    score = xl;                                        // xl dead after k_agg
  unsigned* keys  = (unsigned*)((char*)xl + ((size_t)N_NODES * 4 + 256));
  int*      perm  = (int*)((char*)xl + 2 * ((size_t)N_NODES * 4 + 256));
  float*    vals  = (float*)((char*)xl + 3 * ((size_t)N_NODES * 4 + 256));
  int*      inv   = (int*)((char*)xl + 4 * ((size_t)N_NODES * 4 + 256));
  int*      stain = (int*)((char*)xl + 5 * ((size_t)N_NODES * 4 + 256));
  // small persistent scratch:
  unsigned* st   = (unsigned*)ALLOC(16 * 4);
  int*      hist = (int*)ALLOC(256 * 4);
  float*    wsum = (float*)ALLOC(S_STAINS * 4);
  float*    cntS = (float*)ALLOC(S_STAINS * 4);
  float*    sumv = (float*)ALLOC(4);
  float*    avgS = (float*)ALLOC((size_t)S_STAINS * HID * 4);
  float*    mxS  = (float*)ALLOC((size_t)S_STAINS * HID * 4);

  const int B = 256;
  // --- init graph state (never mutate inputs) ---
  hipMemcpyAsync(srcB, eidx,            (size_t)N_EDGES * 4, hipMemcpyDeviceToDevice, stream);
  hipMemcpyAsync(dstB, eidx + N_EDGES,  (size_t)N_EDGES * 4, hipMemcpyDeviceToDevice, stream);
  k_fill_i<<<CDIV(N_EDGES, B), B, 0, stream>>>(maskB, 1, N_EDGES);
  k_iota  <<<CDIV(N_NODES, B), B, 0, stream>>>(origA, N_NODES);

  int nArr[NLAYERS + 1]   = {30000, 21000, 14700, 10290, 7203};
  int indimArr[NLAYERS]   = {F_IN + WALK, HID + WALK, HID + WALK, HID + WALK};
  int KTarr[NLAYERS]      = {33, 9, 9, 9};           // ceil(indim/32)

  const float* curX = x_in;   int xdim = F_IN;
  const float* curRW = rw_in; float* nxtRW = rwA;
  int* curOrig = origA;       int* nxtOrig = origB;

  for (int i = 0; i < NLAYERS; ++i) {
    int n = nArr[i], k = nArr[i + 1];
    int indim = indimArr[i], KT = KTarr[i];
    int MT = CDIV(n, 16);
    int base = 2 + 9 * i;
    const float* Wl    = (const float*)d_in[base + iWl];
    const float* bl    = (const float*)d_in[base + iBl];
    const float* Wr    = (const float*)d_in[base + iWr];
    const float* br    = (const float*)d_in[base + iBr];
    const float* att   = (const float*)d_in[base + iAtt];
    const float* bias  = (const float*)d_in[base + iBias];
    const float* Wrel  = (const float*)d_in[base + iWrel];
    const float* brel  = (const float*)d_in[base + iBrel];
    const float* Wroot = (const float*)d_in[base + iWroot];
    int E2 = N_EDGES + n;

    // --- WMMA GEMMs: xl = h@Wl+bl, xr = h@Wr+br ---
    k_packA<<<CDIV(MT * KT * 32, B), B, 0, stream>>>(curX, curRW, Apack, n, xdim, indim, MT, KT);
    k_packB<<<CDIV(KT * 1024, B), B, 0, stream>>>(Wl, Bpack, indim, KT);
    if (KT == 33) k_gemm<33><<<MT, 128, 0, stream>>>(Apack, Bpack, bl, xl, n);
    else          k_gemm< 9><<<MT, 128, 0, stream>>>(Apack, Bpack, bl, xl, n);
    k_packB<<<CDIV(KT * 1024, B), B, 0, stream>>>(Wr, Bpack, indim, KT);
    if (KT == 33) k_gemm<33><<<MT, 128, 0, stream>>>(Apack, Bpack, br, xr, n);
    else          k_gemm< 9><<<MT, 128, 0, stream>>>(Apack, Bpack, br, xr, n);

    // --- GATv2 attention + softmax + aggregation ---
    k_fill_f<<<CDIV(n * 2, B), B, 0, stream>>>(mmax, -3.0e38f, n * 2);
    k_logits<<<CDIV(E2 * 32, B), B, 0, stream>>>(xl, xr, att, srcB, dstB, maskB, logit, mmax, N_EDGES, n);
    k_fill_f<<<CDIV(n * 2, B), B, 0, stream>>>(denom, 0.f, n * 2);
    k_exps  <<<CDIV(E2 * 2, B), B, 0, stream>>>(logit, dstB, maskB, mmax, denom, N_EDGES, n);
    k_fill_f<<<CDIV(n * HD, B), B, 0, stream>>>(agg, 0.f, n * HD);
    k_agg   <<<CDIV(E2 * 32, B), B, 0, stream>>>(xl, logit, denom, srcB, dstB, agg, N_EDGES, n);
    k_relumean<<<CDIV(n * HID, B), B, 0, stream>>>(agg, bias, xMid, n);

    // --- SAGPool score ---
    k_fill_f<<<CDIV(n * HID, B), B, 0, stream>>>(msg, 0.f, n * HID);
    k_msg   <<<CDIV(N_EDGES * 32, B), B, 0, stream>>>(xMid, srcB, dstB, maskB, msg, N_EDGES);
    k_score <<<CDIV(n * 32, B), B, 0, stream>>>(xMid, msg, Wrel, brel, Wroot, score, keys, n);

    // --- device-side radix top-k (k known on host) ---
    k_init_state<<<1, 1, 0, stream>>>(st, k);
    for (int shift = 24; shift >= 0; shift -= 8) {
      k_fill_i<<<1, 256, 0, stream>>>(hist, 0, 256);
      k_hist  <<<CDIV(n, B), B, 0, stream>>>(keys, st, hist, n, shift);
      k_scanpick<<<1, 1, 0, stream>>>(st, hist, shift);
    }
    k_compact_gt<<<CDIV(n, B), B, 0, stream>>>(keys, score, st, perm, vals, n);
    k_compact_eq<<<CDIV(n, B), B, 0, stream>>>(keys, score, st, perm, vals, n, k);

    // --- gather selected nodes, remap edges ---
    k_gather_x   <<<CDIV(k * HID, B), B, 0, stream>>>(xMid, perm, vals, xSel, k);
    k_gather_misc<<<CDIV(k, B), B, 0, stream>>>(perm, curRW, curOrig, nattr, nxtRW, nxtOrig, stain, k);
    k_fill_i     <<<CDIV(n, B), B, 0, stream>>>(inv, -1, n);
    k_scatter_inv<<<CDIV(k, B), B, 0, stream>>>(perm, inv, k);
    k_remap      <<<CDIV(N_EDGES, B), B, 0, stream>>>(srcB, dstB, maskB, inv, N_EDGES);

    // --- stain-grouped weighted mean/max pooling -> 512 outputs / layer ---
    k_fill_f<<<1, 64, 0, stream>>>(wsum, 0.f, S_STAINS);
    k_fill_f<<<1, 64, 0, stream>>>(cntS, 0.f, S_STAINS);
    k_fill_f<<<1, 64, 0, stream>>>(sumv, 0.f, 1);
    k_fill_f<<<CDIV(S_STAINS * HID, B), B, 0, stream>>>(avgS, 0.f, S_STAINS * HID);
    k_fill_f<<<CDIV(S_STAINS * HID, B), B, 0, stream>>>(mxS, -3.0e38f, S_STAINS * HID);
    k_pool_scalar<<<CDIV(k, B), B, 0, stream>>>(vals, stain, wsum, cntS, sumv, k);
    k_pool_vec   <<<CDIV(k * HID, B), B, 0, stream>>>(xSel, stain, avgS, mxS, k);
    k_finalize   <<<1, HID, 0, stream>>>(wsum, cntS, avgS, mxS, sumv, (float*)d_out + (size_t)i * 2 * HID);

    // --- rotate ping-pong buffers for next layer ---
    curX = xSel; xdim = HID;
    curRW = nxtRW;   nxtRW   = (nxtRW   == rwA)   ? rwB   : rwA;
    curOrig = nxtOrig; nxtOrig = (nxtOrig == origA) ? origB : origA;
  }
}
